// AttentionDecoder_3487513444504
// MI455X (gfx1250) — matrix operand order repaired and checked by reading, not compiled, not run on hardware
//
#include <hip/hip_runtime.h>

// ---------------------------------------------------------------------------
// AttentionDecoder for MI455X (gfx1250, wave32, WMMA bf16 f32-accum)
// B=4, C=256, C8=32, H=W=64, N=4096
// ---------------------------------------------------------------------------

#define BATCH 4
#define CH    256
#define C8    32
#define NN    4096   // H*W

typedef __attribute__((ext_vector_type(16))) __bf16 v16bf;
typedef __attribute__((ext_vector_type(8)))  __bf16 v8bf;
typedef __attribute__((ext_vector_type(8)))  float  v8f;
typedef __attribute__((ext_vector_type(4)))  int    v4i;

#define WMMA_BF16(A, B, Cc) \
  __builtin_amdgcn_wmma_f32_16x16x32_bf16(false, (A), false, (B), (short)0, (Cc), false, false)

// ---- async Global->LDS copy (ASYNCcnt path), with safe fallback -----------
#if defined(__has_builtin)
#if __has_builtin(__builtin_amdgcn_global_load_async_to_lds_b128)
#define HAVE_ASYNC_LDS 1
#endif
#endif

static __device__ inline void cp16(__bf16* dst_lds, const __bf16* src_glb) {
#ifdef HAVE_ASYNC_LDS
  typedef __attribute__((address_space(1))) v4i* gv4p;   // global int4*
  typedef __attribute__((address_space(3))) v4i* lv4p;   // LDS int4*
  __builtin_amdgcn_global_load_async_to_lds_b128(
      (gv4p)(v4i*)(void*)const_cast<__bf16*>(src_glb),
      (lv4p)(v4i*)(void*)dst_lds, 0, 0);
#else
  *(uint4*)dst_lds = *(const uint4*)src_glb;
#endif
}

static __device__ inline void async_copy_wait() {
#ifdef HAVE_ASYNC_LDS
#if __has_builtin(__builtin_amdgcn_s_wait_asynccnt)
  __builtin_amdgcn_s_wait_asynccnt(0);
#else
  asm volatile("s_wait_asynccnt 0" ::: "memory");
#endif
#endif
}

// Load 16 contiguous f32 and convert to a 16xbf16 fragment half-row.
static __device__ inline v16bf cvt16(const float* __restrict__ p) {
  const float4* q = (const float4*)p;
  float4 a = q[0], b = q[1], c = q[2], d = q[3];
  v16bf r;
  r[0]=(__bf16)a.x; r[1]=(__bf16)a.y; r[2]=(__bf16)a.z; r[3]=(__bf16)a.w;
  r[4]=(__bf16)b.x; r[5]=(__bf16)b.y; r[6]=(__bf16)b.z; r[7]=(__bf16)b.w;
  r[8]=(__bf16)c.x; r[9]=(__bf16)c.y; r[10]=(__bf16)c.z; r[11]=(__bf16)c.w;
  r[12]=(__bf16)d.x; r[13]=(__bf16)d.y; r[14]=(__bf16)d.z; r[15]=(__bf16)d.w;
  return r;
}

// ---------------------------------------------------------------------------
// Kernel A: 1x1-conv projections as WMMA GEMMs.
//   Q  [b][n][c8]  = Wq x  + bq              (A-matrix layout for QK)
//   K' [b][c8][n]  = Wk xe + bk + pos(c8,n)  (B-matrix layout for QK)
//   Vt [b][n][c]   = (Wv xe + bv)^T          (B-matrix layout for PV)
// Grid: 4 batches x 32 n-tiles of 128; 8 waves/block, 16 n-cols per wave.
// ---------------------------------------------------------------------------
__global__ __launch_bounds__(256)
void proj_kernel(const float* __restrict__ x,  const float* __restrict__ xe,
                 const float* __restrict__ Wq, const float* __restrict__ bq,
                 const float* __restrict__ Wk, const float* __restrict__ bk,
                 const float* __restrict__ Wv, const float* __restrict__ bv,
                 const float* __restrict__ h_pos, const float* __restrict__ w_pos,
                 __bf16* __restrict__ Qw, __bf16* __restrict__ Kw,
                 __bf16* __restrict__ Vw)
{
  const int lane = threadIdx.x & 31;
  const int wv   = threadIdx.x >> 5;
  const int b    = blockIdx.x >> 5;
  const int nt   = blockIdx.x & 31;
  const int nb   = nt * 128 + wv * 16;   // wave's 16-column n base
  const int half = lane >> 4;
  const int l16  = lane & 15;
  const int n    = nb + l16;             // this lane's C/D column

  const size_t xbase = (size_t)b * CH * NN;

  // B fragments (K=32 chunks x 8 = C=256) kept resident in VGPRs.
  // B layout used here (contiguous k, same as A): lane L holds column n = L%16, K rows (L/16)*16..+15.
  v16bf Bx[8], Be[8];
#pragma unroll
  for (int kk = 0; kk < 8; ++kk) {
#pragma unroll
    for (int s = 0; s < 16; ++s) {
      Bx[kk][s] = (__bf16)x [xbase + (size_t)(kk * 32 + half * 16 + s) * NN + n];
      Be[kk][s] = (__bf16)xe[xbase + (size_t)(kk * 32 + half * 16 + s) * NN + n];
    }
  }

  // ---- Q = Wq*x + bq, stored [n][c8] (row n contiguous in c8) ----
#pragma unroll
  for (int mt = 0; mt < 2; ++mt) {
    v8f acc;
#pragma unroll
    for (int r = 0; r < 8; ++r) acc[r] = bq[mt * 16 + half * 8 + r];
#pragma unroll
    for (int kk = 0; kk < 8; ++kk) {
      v16bf a = cvt16(Wq + (size_t)(mt * 16 + l16) * CH + kk * 32 + half * 16);
      acc = WMMA_BF16(a, Bx[kk], acc);
    }
    v8bf o;
#pragma unroll
    for (int r = 0; r < 8; ++r) o[r] = (__bf16)acc[r];
    *(v8bf*)(Qw + (size_t)b * NN * C8 + (size_t)n * C8 + mt * 16 + half * 8) = o;
  }

  // ---- K' = Wk*xe + bk + (h_pos + w_pos), stored [c8][n] ----
#pragma unroll
  for (int mt = 0; mt < 2; ++mt) {
    v8f acc;
#pragma unroll
    for (int r = 0; r < 8; ++r) {
      int m = mt * 16 + half * 8 + r;
      acc[r] = bk[m] + h_pos[m * 64 + (n >> 6)] + w_pos[m * 64 + (n & 63)];
    }
#pragma unroll
    for (int kk = 0; kk < 8; ++kk) {
      v16bf a = cvt16(Wk + (size_t)(mt * 16 + l16) * CH + kk * 32 + half * 16);
      acc = WMMA_BF16(a, Be[kk], acc);
    }
#pragma unroll
    for (int r = 0; r < 8; ++r) {
      int m = mt * 16 + half * 8 + r;
      Kw[(size_t)b * C8 * NN + (size_t)m * NN + n] = (__bf16)acc[r];
    }
  }

  // ---- V = Wv*xe + bv, stored transposed [n][c] ----
#pragma unroll
  for (int mt = 0; mt < 16; ++mt) {
    v8f acc;
#pragma unroll
    for (int r = 0; r < 8; ++r) acc[r] = bv[mt * 16 + half * 8 + r];
#pragma unroll
    for (int kk = 0; kk < 8; ++kk) {
      v16bf a = cvt16(Wv + (size_t)(mt * 16 + l16) * CH + kk * 32 + half * 16);
      acc = WMMA_BF16(a, Be[kk], acc);
    }
    v8bf o;
#pragma unroll
    for (int r = 0; r < 8; ++r) o[r] = (__bf16)acc[r];
    *(v8bf*)(Vw + (size_t)b * NN * CH + (size_t)n * CH + mt * 16 + half * 8) = o;
  }
}

// ---------------------------------------------------------------------------
// Kernel B: flash attention, LDS-staged.
// Per block (8 waves, 128 query rows): stream keys in 64-column chunks.
// K'/V chunk staged cooperatively into LDS (async Global->LDS when available),
// all waves read B fragments from LDS (16-bit column reads) -> 8x cut in L2 traffic.
// Per iteration: 4 QK WMMAs + online softmax + 32 PV WMMAs.
// out = gamma * (softmax(Q K') V^T) + x
// ---------------------------------------------------------------------------
__global__ __launch_bounds__(256)
void attn_kernel(const float* __restrict__ x, const float* __restrict__ gamma_p,
                 const __bf16* __restrict__ Qw, const __bf16* __restrict__ Kw,
                 const __bf16* __restrict__ Vw, float* __restrict__ out)
{
  __shared__ __align__(64) __bf16 Kc[C8][64];       //  4 KB  K' chunk [c8][m]
  __shared__ __align__(64) __bf16 Vc[64][CH];       // 32 KB  V  chunk [m][c]
  __shared__ __align__(64) __bf16 Plds[8][16][64];  // 16 KB  per-wave P tile

  const int tid  = threadIdx.x;
  const int lane = tid & 31;
  const int wv   = tid >> 5;
  const int b    = blockIdx.x >> 5;
  const int qt   = blockIdx.x & 31;
  const int half = lane >> 4;
  const int l16  = lane & 15;
  const int nw   = qt * 128 + wv * 16;   // wave's query-row base

  const __bf16* Qb = Qw + (size_t)b * NN * C8;
  const __bf16* Kb = Kw + (size_t)b * C8 * NN;
  const __bf16* Vb = Vw + (size_t)b * NN * CH;

  // Q as A fragment: lane holds row n = nw + l16, K cols half*16..+15.
  v16bf qa = *(const v16bf*)(Qb + (size_t)(nw + l16) * C8 + half * 16);

  v8f O[16];
#pragma unroll
  for (int ct = 0; ct < 16; ++ct)
#pragma unroll
    for (int r = 0; r < 8; ++r) O[ct][r] = 0.f;

  float mrow[8], lrow[8];
#pragma unroll
  for (int r = 0; r < 8; ++r) { mrow[r] = -1e30f; lrow[r] = 0.f; }

  // cooperative-copy coordinates
  const int krow = tid >> 3, kcol = (tid & 7) * 8;   // K: 32x64 = 256 x 16B
  const int vrow = tid >> 2, vcb  = (tid & 3) * 64;  // V: 64x256 = 256 x 128B

  for (int m0 = 0; m0 < NN; m0 += 64) {
    // ---- stage K'/V chunk into LDS (ASYNCcnt path when available) ----
    cp16(&Kc[krow][kcol], Kb + (size_t)krow * NN + m0 + kcol);
#pragma unroll
    for (int i = 0; i < 8; ++i)
      cp16(&Vc[vrow][vcb + i * 8], Vb + (size_t)(m0 + vrow) * CH + vcb + i * 8);
    async_copy_wait();
    __syncthreads();

    // ---- scores: S = Q K' over four 16-col tiles, f32 accum ----
    v8f s[4];
#pragma unroll
    for (int mt = 0; mt < 4; ++mt) {
      v16bf kb;
#pragma unroll
      for (int s = 0; s < 16; ++s) kb[s] = Kc[half * 16 + s][mt * 16 + l16];
      v8f z = {};
      s[mt] = WMMA_BF16(qa, kb, z);
    }

    // ---- online softmax: row stats via 16-lane butterfly (stays in half) --
    float scale[8];
#pragma unroll
    for (int r = 0; r < 8; ++r) {
      float t = fmaxf(fmaxf(s[0][r], s[1][r]), fmaxf(s[2][r], s[3][r]));
#pragma unroll
      for (int off = 8; off >= 1; off >>= 1)
        t = fmaxf(t, __shfl_xor(t, off, 32));
      float mn = fmaxf(mrow[r], t);
      scale[r] = __expf(mrow[r] - mn);
      float sum = 0.f;
#pragma unroll
      for (int mt = 0; mt < 4; ++mt) {
        float p = __expf(s[mt][r] - mn);
        sum += p;
        // D-layout element (row r+8*half, col l16 of tile mt) -> LDS transpose
        Plds[wv][half * 8 + r][mt * 16 + l16] = (__bf16)p;
      }
#pragma unroll
      for (int off = 8; off >= 1; off >>= 1)
        sum += __shfl_xor(sum, off, 32);
      lrow[r] = lrow[r] * scale[r] + sum;
      mrow[r] = mn;
    }
    __builtin_amdgcn_wave_barrier();

    // ---- rescale O once per 64-col chunk ----
#pragma unroll
    for (int ct = 0; ct < 16; ++ct)
#pragma unroll
      for (int r = 0; r < 8; ++r) O[ct][r] *= scale[r];

    // ---- accumulate P V^T: two K=32 sub-chunks x 16 channel tiles ----
#pragma unroll
    for (int mk = 0; mk < 2; ++mk) {
      // P in A-fragment layout: lane row = l16, K cols = mk*32 + half*16..+15
      v16bf pa = *(const v16bf*)(&Plds[wv][l16][mk * 32 + half * 16]);
#pragma unroll
      for (int ct = 0; ct < 16; ++ct) {
        v16bf vb;
#pragma unroll
        for (int s = 0; s < 16; ++s) vb[s] = Vc[mk * 32 + half * 16 + s][ct * 16 + l16];
        O[ct] = WMMA_BF16(pa, vb, O[ct]);
      }
    }
    __syncthreads();   // protect Kc/Vc before next stage
  }

  // ---- epilogue: normalize, gamma, residual ----
  const float g = gamma_p[0];
  float inv[8];
#pragma unroll
  for (int r = 0; r < 8; ++r) inv[r] = 1.f / lrow[r];

#pragma unroll
  for (int ct = 0; ct < 16; ++ct) {
#pragma unroll
    for (int r = 0; r < 8; ++r) {
      int c = ct * 16 + half * 8 + r;
      size_t idx = ((size_t)b * CH + c) * NN + nw + l16;
      out[idx] = g * (O[ct][r] * inv[r]) + x[idx];
    }
  }
}

// ---------------------------------------------------------------------------
extern "C" void kernel_launch(void* const* d_in, const int* in_sizes, int n_in,
                              void* d_out, int out_size, void* d_ws, size_t ws_size,
                              hipStream_t stream) {
  (void)in_sizes; (void)n_in; (void)out_size; (void)ws_size;
  const float* x     = (const float*)d_in[0];
  const float* xe    = (const float*)d_in[1];
  const float* Wq    = (const float*)d_in[2];
  const float* bq    = (const float*)d_in[3];
  const float* Wk    = (const float*)d_in[4];
  const float* bk    = (const float*)d_in[5];
  const float* Wv    = (const float*)d_in[6];
  const float* bv    = (const float*)d_in[7];
  const float* h_pos = (const float*)d_in[8];
  const float* w_pos = (const float*)d_in[9];
  const float* gamma = (const float*)d_in[10];

  __bf16* Qw = (__bf16*)d_ws;                        // B*N*C8   bf16 (1 MB)
  __bf16* Kw = Qw + (size_t)BATCH * NN * C8;         // B*C8*N   bf16 (1 MB)
  __bf16* Vw = Kw + (size_t)BATCH * C8 * NN;         // B*N*C    bf16 (8 MB)

  proj_kernel<<<BATCH * 32, 256, 0, stream>>>(x, xe, Wq, bq, Wk, bk, Wv, bv,
                                              h_pos, w_pos, Qw, Kw, Vw);
  attn_kernel<<<BATCH * 32, 256, 0, stream>>>(x, gamma, Qw, Kw, Vw, (float*)d_out);
}
